// NeuralODE_5257039970814
// MI455X (gfx1250) — compile-verified, matrix operand order checked by reading, not run on hardware
//
#include <hip/hip_runtime.h>

typedef __attribute__((ext_vector_type(16))) _Float16 v16h;
typedef __attribute__((ext_vector_type(8)))  _Float16 v8h;
typedef __attribute__((ext_vector_type(2)))  __fp16   h2;
typedef __attribute__((ext_vector_type(8)))  float    v8f;
typedef __attribute__((ext_vector_type(2)))  float    f2;

namespace {

constexpr int WAVES  = 4;     // waves per block; each wave owns 16 particles
constexpr int BATCH  = 8192;
constexpr int TSTEPS = 64;
constexpr int WID    = 128;

union V16u { v16h v; h2 p[8]; };

__device__ __forceinline__ float capv(float x) { return fminf(fmaxf(x, -5.0f), 5.0f); }

// Packed f16 leaky_relu on a cvt_pk pair: v_pk_mul_f16 + v_pk_max_num_f16.
__device__ __forceinline__ h2 lk2(h2 v) {
    const h2 s = v * (h2){(__fp16)0.01f, (__fp16)0.01f};
    return __builtin_elementwise_max(v, s);
}

// K-index within a 32-wide chunk -> (lane half, f16 element j) per the CDNA5
// 16-bit A/B fragment layout: lanes 0-15 hold K {0-7,16-23}, lanes 16-31 hold
// K {8-15,24-31}; element j<8 covers the first group, j>=8 the second.
__device__ __forceinline__ void kmap(int kr, int& lhi, int& j) {
    if      (kr <  8) { lhi = 0; j = kr;            }
    else if (kr < 16) { lhi = 1; j = kr - 8;        }
    else if (kr < 24) { lhi = 0; j = 8 + (kr - 16); }
    else              { lhi = 1; j = 8 + (kr - 24); }
}

// Fragments stored as [frag][jhalf][lane][8] halves: per-lane stride 16B ->
// conflict-free ds_load_b128 (lanes 0-15 touch each bank exactly once).
__device__ __forceinline__ int frag_dst(int f, int lane, int j) {
    return f * 512 + ((j >= 8) ? 256 : 0) + lane * 8 + (j & 7);
}

// loff = laundered (lane*8) so these loads are NOT loop-invariant and stay
// inside the step loop as ds_load_b128 pairs (no hoist -> no scratch spill).
__device__ __forceinline__ v16h ldfrag(const _Float16* base, int f, int loff) {
    const _Float16* p = base + f * 512 + loff;
    const v8h lo = *(const v8h*)p;
    const v8h hi = *(const v8h*)(p + 256);
    v16h a;
#pragma unroll
    for (int j = 0; j < 8; ++j) { a[j] = lo[j]; a[8 + j] = hi[j]; }
    return a;
}

// One 128->128 layer: Bout = leaky(W @ Bin + b), all activations in registers.
// Processes output tiles in pairs (2kc, 2kc+1): the pair's 8 weight fragments
// are preloaded into distinct registers (16 back-to-back ds_load_b128) so the
// 8 WMMAs overlap the loads with staged dscnt waits, and the pair's D
// accumulators pack directly into next-layer B-fragment kc.
__device__ __forceinline__ void dense128(const _Float16* wf, const float* bs,
                                         const v16h* Bin, v16h* Bout,
                                         int loff, int bhalf)
{
#pragma unroll
    for (int kc = 0; kc < 4; ++kc) {
        v16h a[8];
#pragma unroll
        for (int f = 0; f < 8; ++f) a[f] = ldfrag(wf, (2 * kc) * 4 + f, loff);
        v8f c0 = *(const v8f*)(bs + (2 * kc) * 16 + bhalf);
        v8f c1 = *(const v8f*)(bs + (2 * kc + 1) * 16 + bhalf);
#pragma unroll
        for (int k2 = 0; k2 < 4; ++k2)
            c0 = __builtin_amdgcn_wmma_f32_16x16x32_f16(false, a[k2], false, Bin[k2],
                                                        (short)0, c0, false, false);
#pragma unroll
        for (int k2 = 0; k2 < 4; ++k2)
            c1 = __builtin_amdgcn_wmma_f32_16x16x32_f16(false, a[4 + k2], false, Bin[k2],
                                                        (short)0, c1, false, false);
        V16u b;
#pragma unroll
        for (int q = 0; q < 4; ++q) {
            b.p[q]     = lk2(__builtin_amdgcn_cvt_pkrtz(c0[2 * q], c0[2 * q + 1]));
            b.p[4 + q] = lk2(__builtin_amdgcn_cvt_pkrtz(c1[2 * q], c1[2 * q + 1]));
        }
        Bout[kc] = b.v;
    }
}

__global__ void __launch_bounds__(WAVES * 32, 4)
__attribute__((amdgpu_waves_per_eu(4)))
neural_ode_wmma(const float* __restrict__ ts, const float* __restrict__ y0,
                const float* __restrict__ W1, const float* __restrict__ b1,
                const float* __restrict__ W2, const float* __restrict__ b2,
                const float* __restrict__ W3, const float* __restrict__ b3,
                const float* __restrict__ W4, const float* __restrict__ b4,
                float* __restrict__ out)
{
    extern __shared__ __attribute__((aligned(32))) char smem[];
    _Float16* w1f = (_Float16*)smem;        // 8 frags  (A = W1|b1 padded to K=32)
    _Float16* w2f = w1f + 8 * 512;          // 32 frags (A = W2, 8 ntile x 4 kc)
    _Float16* w3f = w2f + 32 * 512;         // 32 frags
    _Float16* w4f = w3f + 32 * 512;         // 4 frags  (A = W4, N padded 2->16)
    float* b2s = (float*)(w4f + 4 * 512);
    float* b3s = b2s + WID;
    float* b4s = b3s + WID;                 // 2
    float* tss = b4s + 2;                   // TSTEPS

    const int tid = threadIdx.x;

    // ---- one-time staging of weights into WMMA A-fragment layout ----
    for (int idx = tid; idx < WID * WID; idx += blockDim.x) {
        const int n = idx >> 7, k = idx & 127;           // n = out feature (A row)
        const int kc = k >> 5, kr = k & 31;
        int lhi, j; kmap(kr, lhi, j);
        const int lane = lhi * 16 + (n & 15);
        const int dst  = frag_dst(((n >> 4) * 4) + kc, lane, j);
        w2f[dst] = (_Float16)W2[n * WID + k];            // A(n,k) = W2[n][k]
        w3f[dst] = (_Float16)W3[n * WID + k];
    }
    for (int idx = tid; idx < WID * 32; idx += blockDim.x) {
        const int n = idx >> 5, kr = idx & 31;
        int lhi, j; kmap(kr, lhi, j);
        float v = 0.0f;
        if      (kr == 0) v = W1[n * 2 + 0];
        else if (kr == 1) v = W1[n * 2 + 1];
        else if (kr == 2) v = b1[n];                     // bias row, pairs with B's constant 1
        w1f[frag_dst(n >> 4, lhi * 16 + (n & 15), j)] = (_Float16)v;
    }
    for (int idx = tid; idx < 16 * WID; idx += blockDim.x) {
        const int n = idx >> 7, k = idx & 127;
        const int kc = k >> 5, kr = k & 31;
        int lhi, j; kmap(kr, lhi, j);
        w4f[frag_dst(kc, lhi * 16 + (n & 15), j)] =
            (n < 2) ? (_Float16)W4[n * WID + k] : (_Float16)0.0f;
    }
    for (int idx = tid; idx < WID; idx += blockDim.x) { b2s[idx] = b2[idx]; b3s[idx] = b3[idx]; }
    for (int idx = tid; idx < TSTEPS; idx += blockDim.x) tss[idx] = ts[idx];
    if (tid < 2) b4s[tid] = b4[tid];
    __syncthreads();

    const int wave  = tid >> 5;
    const int lane  = tid & 31;
    const int mhalf = (lane >> 4) << 3;                  // C/D row offset: 0 or 8
    const int p0    = (blockIdx.x * WAVES + wave) * 16;  // particle base of this wave
    const bool own  = lane < 16;                         // lane m owns particle p0+m

    // ---- per-lane resident state: y of the owned particle ----
    float y0r = 0.0f, y1r = 0.0f;
    if (own) {
        const f2 yy = *(const f2*)(y0 + (p0 + lane) * 2);
        y0r = yy.x; y1r = yy.y;
        f2 o; o.x = capv(y0r); o.y = capv(y1r);
        *(f2*)(out + (p0 + lane) * 2) = o;               // ys[0] = clip(y0)
    }
    const float b40 = b4s[0], b41 = b4s[1];
    const _Float16 h1c = (_Float16)1.0f;
    const _Float16 h0c = (_Float16)0.0f;

    for (int s = 0; s < TSTEPS - 1; ++s) {
        const float dt = tss[s + 1] - tss[s];

        // Launder per-lane offsets each iteration: breaks loop-invariance of
        // the weight/bias LDS loads so they are re-issued from LDS every step
        // instead of being hoisted into (and spilled out of) registers.
        int loff  = lane * 8;                            // halves
        int bhalf = mhalf;                               // floats
        asm volatile("" : "+v"(loff), "+v"(bhalf));

        // ---- layer 1: B = [y0; y1; 1; 0...] (K padded to 32), A = staged W1|b1 ----
        v16h bIn = {};
        bIn[0] = own ? (_Float16)y0r : h0c;
        bIn[1] = own ? (_Float16)y1r : h0c;
        bIn[2] = own ? h1c : h0c;

        v16h Bh[4], Bh2[4];
        {
            v16h a1[8];
#pragma unroll
            for (int t = 0; t < 8; ++t) a1[t] = ldfrag(w1f, t, loff);
#pragma unroll
            for (int kc = 0; kc < 4; ++kc) {
                v8f c0 = {}, c1 = {};
                c0 = __builtin_amdgcn_wmma_f32_16x16x32_f16(false, a1[2 * kc], false, bIn,
                                                            (short)0, c0, false, false);
                c1 = __builtin_amdgcn_wmma_f32_16x16x32_f16(false, a1[2 * kc + 1], false, bIn,
                                                            (short)0, c1, false, false);
                V16u b;
#pragma unroll
                for (int q = 0; q < 4; ++q) {
                    b.p[q]     = lk2(__builtin_amdgcn_cvt_pkrtz(c0[2 * q], c0[2 * q + 1]));
                    b.p[4 + q] = lk2(__builtin_amdgcn_cvt_pkrtz(c1[2 * q], c1[2 * q + 1]));
                }
                Bh[kc] = b.v;
            }
        }

        // ---- layers 2 and 3: activations never leave registers ----
        dense128(w2f, b2s, Bh,  Bh2, loff, bhalf);
        dense128(w3f, b3s, Bh2, Bh,  loff, bhalf);

        // ---- layer 4: f = W4 @ h3 + b4 (rows padded 2->16) ----
        v8f c4 = {};
        c4[0] = own ? b40 : 0.0f;
        c4[1] = own ? b41 : 0.0f;
        {
            v16h a4[4];
#pragma unroll
            for (int kc = 0; kc < 4; ++kc) a4[kc] = ldfrag(w4f, kc, loff);
#pragma unroll
            for (int kc = 0; kc < 4; ++kc)
                c4 = __builtin_amdgcn_wmma_f32_16x16x32_f16(false, a4[kc], false, Bh[kc],
                                                            (short)0, c4, false, false);
        }

        // ---- Euler update + emit ys[s+1], all within the owning lane ----
        if (own) {
            y0r = y0r + dt * (c4[0] - y0r);              // y + dt*(-y + f)
            y1r = y1r + dt * (c4[1] - y1r);
            f2 o; o.x = capv(y0r); o.y = capv(y1r);
            *(f2*)(out + (s + 1) * (BATCH * 2) + (p0 + lane) * 2) = o;
        }
    }
}

} // namespace

extern "C" void kernel_launch(void* const* d_in, const int* in_sizes, int n_in,
                              void* d_out, int out_size, void* d_ws, size_t ws_size,
                              hipStream_t stream)
{
    (void)in_sizes; (void)n_in; (void)out_size; (void)d_ws; (void)ws_size;
    const float* ts = (const float*)d_in[0];
    const float* y0 = (const float*)d_in[1];
    const float* W1 = (const float*)d_in[2];
    const float* b1 = (const float*)d_in[3];
    const float* W2 = (const float*)d_in[4];
    const float* b2 = (const float*)d_in[5];
    const float* W3 = (const float*)d_in[6];
    const float* b3 = (const float*)d_in[7];
    const float* W4 = (const float*)d_in[8];
    const float* b4 = (const float*)d_in[9];
    float* out = (float*)d_out;

    const size_t halves = (size_t)(8 + 32 + 32 + 4) * 512;   // weight fragments
    const size_t floats = 128 + 128 + 2 + TSTEPS;
    const size_t shmem  = halves * 2 + floats * 4;           // ~79 KB

    dim3 grid(BATCH / (WAVES * 16));
    dim3 block(WAVES * 32);
    hipLaunchKernelGGL(neural_ode_wmma, grid, block, shmem, stream,
                       ts, y0, W1, b1, W2, b2, W3, b3, W4, b4, out);
}
